// AdvancedHypergraphNetwork_69801808494761
// MI455X (gfx1250) — matrix operand order repaired and checked by reading, not compiled, not run on hardware
//
#include <hip/hip_runtime.h>
#include <hip/hip_bf16.h>
#include <stdint.h>

#define N_    8192
#define E_    8192
#define NNZ_  262144
#define D_    128
#define H_    4
#define HD_   32
#define L_    3
#define EPS_  1e-05f
#define SLOPE_ 0.2f

typedef __attribute__((ext_vector_type(16))) _Float16 v16h;
typedef __attribute__((ext_vector_type(8)))  float    v8f;

#if defined(__has_builtin)
#  if __has_builtin(__builtin_amdgcn_tensor_load_to_lds)
#    define HAVE_TDM 1
#  endif
#endif
#ifndef HAVE_TDM
#  define HAVE_TDM 0
#endif

#if HAVE_TDM
typedef unsigned int u32x4 __attribute__((ext_vector_type(4)));
typedef int          i32x4 __attribute__((ext_vector_type(4)));
typedef int          i32x8 __attribute__((ext_vector_type(8)));

// Issue one TDM 2-D tile load (f16 elements) global -> LDS.
// D# packing per CDNA5 ISA 8.3/8.4: group0 = {flags, lds_addr, global_addr, type=2},
// group1 = {data_size=2B, tensor_dim0/1, tile_dim0/1, tensor_dim0_stride}.
__device__ __forceinline__ void tdm_load_tile_f16(uint32_t lds_byte_off,
                                                  const _Float16* gsrc,
                                                  uint32_t tensor_w, uint32_t tensor_h,
                                                  uint32_t tile_w, uint32_t tile_h,
                                                  uint32_t stride_elems) {
  uint64_t ga = (uint64_t)(uintptr_t)gsrc;
  u32x4 g0;
  g0.x = 1u;                                                  // count=1, user D#
  g0.y = lds_byte_off;                                        // lds_addr
  g0.z = (uint32_t)ga;                                        // global_addr[95:64]
  g0.w = (uint32_t)((ga >> 32) & 0x01FFFFFFu) | (2u << 30);   // addr[120:96], type=2
  i32x8 g1;
  g1[0] = (int)(1u << 16);                                    // data_size = 1 -> 2 bytes
  g1[1] = (int)((tensor_w & 0xFFFFu) << 16);                  // tensor_dim0[15:0]
  g1[2] = (int)(((tensor_w >> 16) & 0xFFFFu) | ((tensor_h & 0xFFFFu) << 16));
  g1[3] = (int)(((tensor_h >> 16) & 0xFFFFu) | ((tile_w & 0xFFFFu) << 16));
  g1[4] = (int)(tile_h & 0xFFFFu);                            // tile_dim1 (tile_dim2=0)
  g1[5] = (int)stride_elems;                                  // tensor_dim0_stride lo32
  g1[6] = 0;
  g1[7] = 0;
  i32x4 gz = {0, 0, 0, 0};
#if __clang_major__ >= 23
  i32x8 gz8 = {0, 0, 0, 0, 0, 0, 0, 0};
  __builtin_amdgcn_tensor_load_to_lds(g0, g1, gz, gz, gz8, 0);
#else
  __builtin_amdgcn_tensor_load_to_lds(g0, g1, gz, gz, 0);
#endif
}
#endif  // HAVE_TDM

// ---------------------------------------------------------------------------
// WMMA fragment loaders (16x16x32 f16, contiguous k order chosen by this code, wave32)
// A (16xK f16, row-major): lane L holds row M=L&15.
//   lanes 0-15 : v[0..15] = K 0..15
//   lanes 16-31: v[0..15] = K 16..31
// ---------------------------------------------------------------------------
__device__ __forceinline__ v16h load_a_frag(const _Float16* __restrict__ A, int ld,
                                            int row_base, int kb, int lane) {
  int m    = row_base + (lane & 15);
  int off8 = (lane >> 4) * 16;
  const _Float16* p = A + (size_t)m * ld + kb + off8;
  v16h a;
#pragma unroll
  for (int e = 0; e < 8; ++e) a[e] = p[e];
#pragma unroll
  for (int e = 0; e < 8; ++e) a[8 + e] = p[8 + e];
  return a;
}

// B (KxN) taken as W^T where W is [Nout,K] row-major: B[k][n] = W[n][k].
//   lane L holds column n=L&15; lanes 0-15 K=kb..kb+15, lanes 16-31 K=kb+16..kb+31
__device__ __forceinline__ v16h load_b_frag_wt(const _Float16* __restrict__ W, int ld,
                                               int col_base, int kb, int lane) {
  int n     = col_base + (lane & 15);
  int off16 = (lane >> 4) * 16;
  const _Float16* p = W + (size_t)n * ld + kb + off16;
  v16h b;
#pragma unroll
  for (int e = 0; e < 16; ++e) b[e] = p[e];
  return b;
}

// ---------------------------------------------------------------------------
// Generic WMMA GEMM: C[M,Nout] = A[M,K](f16) * W[Nout,K](f16)^T + bias, opt ReLU
// One wave (32 threads) per 16x16 output tile.
// ---------------------------------------------------------------------------
__global__ void wmma_gemm_bias(const _Float16* __restrict__ A,
                               const _Float16* __restrict__ W,
                               const float* __restrict__ bias,
                               float* __restrict__ C,
                               int M, int Nout, int K, int act) {
  int ntn  = Nout >> 4;
  int tile = blockIdx.x;
  int tm   = (tile / ntn) << 4;
  int tn   = (tile % ntn) << 4;
  int lane = threadIdx.x;

  v8f acc = {};
  for (int kb = 0; kb < K; kb += 32) {
    if (kb + 32 < K)
      __builtin_prefetch((const void*)(A + (size_t)(tm + (lane & 15)) * K + kb + 32), 0, 1);
    v16h a = load_a_frag(A, K, tm, kb, lane);
    v16h b = load_b_frag_wt(W, K, tn, kb, lane);
    acc = __builtin_amdgcn_wmma_f32_16x16x32_f16(false, a, false, b, (short)0, acc, false, false);
  }
  int   n  = tn + (lane & 15);
  float bv = bias ? bias[n] : 0.0f;
  int rbase = tm + ((lane >> 4) << 3);
#pragma unroll
  for (int r = 0; r < 8; ++r) {
    float v = acc[r] + bv;
    if (act == 1) v = fmaxf(v, 0.0f);
    C[(size_t)(rbase + r) * Nout + n] = v;
  }
}

// ---------------------------------------------------------------------------
// Flash attention v2: one wave handles 16 query rows of one head, 32 keys per
// iteration. 32x32 K/V tiles are staged in LDS by the Tensor Data Mover
// (TENSORcnt-tracked DMA); S = Q K^T via two WMMAs; online softmax over both
// column halves with half-wave shuffles; full 16x32 P fragment feeds two
// full-K P*V WMMAs (V transposed for free out of LDS).
// ---------------------------------------------------------------------------
__global__ void flash_attn_kernel(const _Float16* __restrict__ Qh,
                                  const _Float16* __restrict__ Kh,
                                  const _Float16* __restrict__ Vh,
                                  float* __restrict__ O) {
  // declaration order fixes LDS offsets: ldsK @ 0, ldsV @ 2048, lds_p @ 4096
  __shared__ _Float16 ldsK[32 * 32];
  __shared__ _Float16 ldsV[32 * 32];
  __shared__ float    lds_p[16 * 32];

  const int   lane  = threadIdx.x;
  const int   lo    = lane & 15;
  const int   hi    = lane >> 4;
  const int   h     = blockIdx.y;
  const int   qbase = blockIdx.x << 4;
  const float scale = 0.17677669529663687f;  // 1/sqrt(32)

  // Q fragment (constant over the key loop)
  const _Float16* qp = Qh + (size_t)(qbase + lo) * D_ + h * HD_ + hi * 16;
  v16h qa;
#pragma unroll
  for (int e = 0; e < 8; ++e) qa[e] = qp[e];
#pragma unroll
  for (int e = 0; e < 8; ++e) qa[8 + e] = qp[8 + e];

  float mrow[8], lrow[8];
  v8f o0 = {}, o1 = {};
#pragma unroll
  for (int r = 0; r < 8; ++r) { mrow[r] = -3.0e38f; lrow[r] = 0.0f; }

  for (int jb = 0; jb < N_; jb += 32) {
#if HAVE_TDM
    // DMA the 32x32 f16 K and V tiles (row stride D_) into LDS via TDM.
    tdm_load_tile_f16(0u,    Kh + (size_t)jb * D_ + h * HD_, D_, N_, 32, 32, D_);
    tdm_load_tile_f16(2048u, Vh + (size_t)jb * D_ + h * HD_, D_, N_, 32, 32, D_);
#  if __has_builtin(__builtin_amdgcn_s_wait_tensorcnt)
    __builtin_amdgcn_s_wait_tensorcnt(0);
#  endif
    __syncthreads();
#else
    // fallback: each lane copies one 64B row of K and V
    {
      const _Float16* kr = Kh + (size_t)(jb + lane) * D_ + h * HD_;
      const _Float16* vr = Vh + (size_t)(jb + lane) * D_ + h * HD_;
#pragma unroll
      for (int e = 0; e < 32; ++e) { ldsK[lane * 32 + e] = kr[e]; ldsV[lane * 32 + e] = vr[e]; }
    }
    __syncthreads();
#endif

    // S tiles: B[k=d][n=j] = Ktile[j][d]; two column halves t=0,1
    v16h kb0, kb1;
#pragma unroll
    for (int e = 0; e < 16; ++e) kb0[e] = ldsK[lo * 32 + hi * 16 + e];
#pragma unroll
    for (int e = 0; e < 16; ++e) kb1[e] = ldsK[(16 + lo) * 32 + hi * 16 + e];
    v8f s0 = {}, s1 = {};
    s0 = __builtin_amdgcn_wmma_f32_16x16x32_f16(false, qa, false, kb0, (short)0, s0, false, false);
    s1 = __builtin_amdgcn_wmma_f32_16x16x32_f16(false, qa, false, kb1, (short)0, s1, false, false);

    // online softmax over 32 columns
#pragma unroll
    for (int r = 0; r < 8; ++r) {
      float v0 = s0[r] * scale;
      float v1 = s1[r] * scale;
      float mx = fmaxf(v0, v1);
      mx = fmaxf(mx, __shfl_xor(mx, 1, 32));
      mx = fmaxf(mx, __shfl_xor(mx, 2, 32));
      mx = fmaxf(mx, __shfl_xor(mx, 4, 32));
      mx = fmaxf(mx, __shfl_xor(mx, 8, 32));
      float mnew = fmaxf(mrow[r], mx);
      float p0   = __expf(v0 - mnew);
      float p1   = __expf(v1 - mnew);
      float rs   = p0 + p1;
      rs += __shfl_xor(rs, 1, 32);
      rs += __shfl_xor(rs, 2, 32);
      rs += __shfl_xor(rs, 4, 32);
      rs += __shfl_xor(rs, 8, 32);
      float corr = __expf(mrow[r] - mnew);
      lrow[r] = lrow[r] * corr + rs;
      mrow[r] = mnew;
      o0[r] *= corr;
      o1[r] *= corr;
      int row = r + hi * 8;
      lds_p[row * 32 + lo]      = p0;
      lds_p[row * 32 + 16 + lo] = p1;
    }
    __syncthreads();

    // full 16x32 P fragment (A layout)
    v16h pa;
#pragma unroll
    for (int e = 0; e < 8; ++e) pa[e] = (_Float16)lds_p[lo * 32 + e + hi * 16];
#pragma unroll
    for (int e = 0; e < 8; ++e) pa[8 + e] = (_Float16)lds_p[lo * 32 + e + 8 + hi * 16];

    // O += P * V : B[k=j][n=d] = Vtile[j][d], transposed out of LDS
#pragma unroll
    for (int c = 0; c < 2; ++c) {
      v16h vb;
#pragma unroll
      for (int e = 0; e < 16; ++e) vb[e] = ldsV[(hi * 16 + e) * 32 + c * 16 + lo];
      if (c == 0)
        o0 = __builtin_amdgcn_wmma_f32_16x16x32_f16(false, pa, false, vb, (short)0, o0, false, false);
      else
        o1 = __builtin_amdgcn_wmma_f32_16x16x32_f16(false, pa, false, vb, (short)0, o1, false, false);
    }
    __syncthreads();
  }
#pragma unroll
  for (int r = 0; r < 8; ++r) {
    int   row = qbase + r + hi * 8;
    float inv = 1.0f / lrow[r];
    O[(size_t)row * D_ + h * HD_ + lo]      = o0[r] * inv;
    O[(size_t)row * D_ + h * HD_ + 16 + lo] = o1[r] * inv;
  }
}

// ---------------------------------------------------------------------------
// Elementwise / scatter helpers
// ---------------------------------------------------------------------------
__global__ void gather_embed(const float* __restrict__ emb, const int* __restrict__ idx,
                             _Float16* __restrict__ xh) {
  int n = blockIdx.x, d = threadIdx.x;
  xh[(size_t)n * D_ + d] = (_Float16)emb[(size_t)idx[n] * D_ + d];
}

__global__ void cvt_f16(const float* __restrict__ a, _Float16* __restrict__ b, size_t n) {
  size_t i = blockIdx.x * (size_t)blockDim.x + threadIdx.x;
  if (i < n) b[i] = (_Float16)a[i];
}

__global__ void fill_f32(float* __restrict__ p, float v, size_t n) {
  size_t i = blockIdx.x * (size_t)blockDim.x + threadIdx.x;
  if (i < n) p[i] = v;
}

__global__ void split_qkv_h(const float* __restrict__ qkv, _Float16* __restrict__ qh,
                            _Float16* __restrict__ kh, _Float16* __restrict__ vh) {
  size_t i = blockIdx.x * (size_t)blockDim.x + threadIdx.x;
  if (i >= (size_t)N_ * D_) return;
  size_t n = i >> 7;
  int    d = (int)(i & 127);
  const float* row = qkv + n * (3 * D_);
  qh[i] = (_Float16)row[d];
  kh[i] = (_Float16)row[D_ + d];
  vh[i] = (_Float16)row[2 * D_ + d];
}

__global__ void compute_deg(const int* __restrict__ edge_idx, float* __restrict__ deg) {
  size_t i = blockIdx.x * (size_t)blockDim.x + threadIdx.x;
  if (i < NNZ_) atomicAdd(&deg[edge_idx[i]], 1.0f);
}

__global__ void recip_pos(const float* __restrict__ a, float* __restrict__ b, int n) {
  int i = blockIdx.x * blockDim.x + threadIdx.x;
  if (i < n) { float v = a[i]; b[i] = (v > 0.0f) ? 1.0f / v : 0.0f; }
}

__global__ void scatter_edge_sum(const float* __restrict__ xl, const int* __restrict__ node_idx,
                                 const int* __restrict__ edge_idx, float* __restrict__ e_attr) {
  int nz = blockIdx.x, d = threadIdx.x;
  atomicAdd(&e_attr[(size_t)edge_idx[nz] * D_ + d], xl[(size_t)node_idx[nz] * D_ + d]);
}

__global__ void scale_rows(float* __restrict__ m, const float* __restrict__ s, size_t total) {
  size_t i = blockIdx.x * (size_t)blockDim.x + threadIdx.x;
  if (i < total) m[i] *= s[i >> 7];
}

__global__ void alpha_kernel(const float* __restrict__ xl, const float* __restrict__ e_attr,
                             const int* __restrict__ node_idx, const int* __restrict__ edge_idx,
                             const float* __restrict__ a_src, const float* __restrict__ a_dst,
                             float* __restrict__ alpha) {
  int wid  = blockIdx.x * (blockDim.x >> 5) + (threadIdx.x >> 5);
  int lane = threadIdx.x & 31;
  if (wid >= NNZ_) return;
  int   n = node_idx[wid], e = edge_idx[wid];
  float s = 0.0f;
  for (int d = lane; d < D_; d += 32)
    s += xl[(size_t)n * D_ + d] * a_src[d] + e_attr[(size_t)e * D_ + d] * a_dst[d];
  s += __shfl_xor(s, 1, 32);
  s += __shfl_xor(s, 2, 32);
  s += __shfl_xor(s, 4, 32);
  s += __shfl_xor(s, 8, 32);
  s += __shfl_xor(s, 16, 32);
  if (lane == 0) alpha[wid] = (s > 0.0f) ? s : SLOPE_ * s;
}

__device__ __forceinline__ void atomicMaxF(float* addr, float val) {
  int* ia  = (int*)addr;
  int  old = *ia;
  while (__int_as_float(old) < val) {
    int prev = atomicCAS(ia, old, __float_as_int(val));
    if (prev == old) break;
    old = prev;
  }
}

__global__ void seg_max_kernel(const float* __restrict__ alpha, const int* __restrict__ node_idx,
                               float* __restrict__ m) {
  size_t i = blockIdx.x * (size_t)blockDim.x + threadIdx.x;
  if (i < NNZ_) atomicMaxF(&m[node_idx[i]], alpha[i]);
}

__global__ void fix_m(float* __restrict__ m, int n) {
  int i = blockIdx.x * blockDim.x + threadIdx.x;
  if (i < n) { float v = m[i]; if (!isfinite(v)) m[i] = 0.0f; }
}

__global__ void ex_kernel(const float* __restrict__ alpha, const float* __restrict__ m,
                          const int* __restrict__ node_idx, float* __restrict__ ex,
                          float* __restrict__ ssum) {
  size_t i = blockIdx.x * (size_t)blockDim.x + threadIdx.x;
  if (i >= NNZ_) return;
  int   n = node_idx[i];
  float e = __expf(alpha[i] - m[n]);
  ex[i] = e;
  atomicAdd(&ssum[n], e);
}

__global__ void norm_alpha(const float* __restrict__ ex, const float* __restrict__ ssum,
                           const int* __restrict__ node_idx, float* __restrict__ alpha) {
  size_t i = blockIdx.x * (size_t)blockDim.x + threadIdx.x;
  if (i >= NNZ_) return;
  float s = ssum[node_idx[i]];
  alpha[i] = ex[i] / ((s > 0.0f) ? s : 1.0f);
}

__global__ void dw_kernel(const float* __restrict__ hw, const int* __restrict__ node_idx,
                          const int* __restrict__ edge_idx, float* __restrict__ Dw) {
  size_t i = blockIdx.x * (size_t)blockDim.x + threadIdx.x;
  if (i < NNZ_) atomicAdd(&Dw[node_idx[i]], hw[edge_idx[i]]);
}

__global__ void scatter_msg1(const float* __restrict__ xl, const float* __restrict__ alpha,
                             const float* __restrict__ Binv, const int* __restrict__ node_idx,
                             const int* __restrict__ edge_idx, float* __restrict__ ef) {
  int nz = blockIdx.x, d = threadIdx.x;
  int e  = edge_idx[nz];
  float c = Binv[e] * alpha[nz];
  atomicAdd(&ef[(size_t)e * D_ + d], c * xl[(size_t)node_idx[nz] * D_ + d]);
}

__global__ void scatter_msg2(const float* __restrict__ ef, const float* __restrict__ alpha,
                             const float* __restrict__ Dinv, const int* __restrict__ node_idx,
                             const int* __restrict__ edge_idx, float* __restrict__ outb) {
  int nz = blockIdx.x, d = threadIdx.x;
  int n  = node_idx[nz];
  float c = Dinv[n] * alpha[nz];
  atomicAdd(&outb[(size_t)n * D_ + d], c * ef[(size_t)edge_idx[nz] * D_ + d]);
}

__global__ void finish_layer(const float* __restrict__ outb, const float* __restrict__ bias,
                             float* __restrict__ x, _Float16* __restrict__ xh) {
  size_t i = blockIdx.x * (size_t)blockDim.x + threadIdx.x;
  if (i >= (size_t)N_ * D_) return;
  float v = fmaxf(outb[i] + bias[i & 127], 0.0f);
  x[i]  = v;
  xh[i] = (_Float16)v;
}

__global__ void hw_kernel(const float* __restrict__ y1, const float* __restrict__ w2,
                          const float* __restrict__ b2, float* __restrict__ hw) {
  int wid  = blockIdx.x * (blockDim.x >> 5) + (threadIdx.x >> 5);
  int lane = threadIdx.x & 31;
  if (wid >= N_) return;
  float s = 0.0f;
  for (int d = lane; d < D_; d += 32) s += y1[(size_t)wid * D_ + d] * w2[d];
  s += __shfl_xor(s, 1, 32);
  s += __shfl_xor(s, 2, 32);
  s += __shfl_xor(s, 4, 32);
  s += __shfl_xor(s, 8, 32);
  s += __shfl_xor(s, 16, 32);
  if (lane == 0) {
    float z = s + b2[0];
    hw[wid] = 1.0f / (1.0f + __expf(-z));
  }
}

__global__ void bn_stats(const float* __restrict__ h, float* __restrict__ mu,
                         float* __restrict__ var) {
  __shared__ float ss[256], ss2[256];
  int   c = blockIdx.x;
  float s = 0.0f, s2 = 0.0f;
  for (int n = threadIdx.x; n < N_; n += 256) {
    float v = h[(size_t)n * 64 + c];
    s += v; s2 += v * v;
  }
  ss[threadIdx.x] = s; ss2[threadIdx.x] = s2;
  __syncthreads();
  for (int st = 128; st > 0; st >>= 1) {
    if (threadIdx.x < st) { ss[threadIdx.x] += ss[threadIdx.x + st]; ss2[threadIdx.x] += ss2[threadIdx.x + st]; }
    __syncthreads();
  }
  if (threadIdx.x == 0) {
    float m = ss[0] / (float)N_;
    mu[c]  = m;
    var[c] = ss2[0] / (float)N_ - m * m;
  }
}

__global__ void bn_apply(const float* __restrict__ h, const float* __restrict__ mu,
                         const float* __restrict__ var, const float* __restrict__ gamma,
                         const float* __restrict__ beta, _Float16* __restrict__ hh) {
  size_t i = blockIdx.x * (size_t)blockDim.x + threadIdx.x;
  if (i >= (size_t)N_ * 64) return;
  int   c = (int)(i & 63);
  float v = gamma[c] * (h[i] - mu[c]) * rsqrtf(var[c] + EPS_) + beta[c];
  hh[i] = (_Float16)fmaxf(v, 0.0f);
}

// ---------------------------------------------------------------------------
// Host orchestration
// ---------------------------------------------------------------------------
static inline int nb(size_t n, int bs) { return (int)((n + (size_t)bs - 1) / (size_t)bs); }

extern "C" void kernel_launch(void* const* d_in, const int* in_sizes, int n_in,
                              void* d_out, int out_size, void* d_ws, size_t ws_size,
                              hipStream_t stream) {
  (void)in_sizes; (void)n_in; (void)out_size; (void)ws_size;

  const int*   kw        = (const int*)d_in[0];
  const int*   he        = (const int*)d_in[1];
  const int*   node_idx  = he;
  const int*   edge_idx  = he + NNZ_;
  const float* emb       = (const float*)d_in[2];
  const float* in_w      = (const float*)d_in[3];
  const float* in_b      = (const float*)d_in[4];
  const float* out_w     = (const float*)d_in[5];
  const float* out_b     = (const float*)d_in[6];
  const float* conv_w    = (const float*)d_in[7];
  const float* conv_b    = (const float*)d_in[8];
  const float* conv_att  = (const float*)d_in[9];
  const float* wg_w1     = (const float*)d_in[10];
  const float* wg_b1     = (const float*)d_in[11];
  const float* wg_w2     = (const float*)d_in[12];
  const float* wg_b2     = (const float*)d_in[13];
  const float* fl_w1     = (const float*)d_in[14];
  const float* fl_b1     = (const float*)d_in[15];
  const float* bn_gamma  = (const float*)d_in[16];
  const float* bn_beta   = (const float*)d_in[17];
  const float* fl_w2     = (const float*)d_in[18];
  const float* fl_b2     = (const float*)d_in[19];
  float*       outp      = (float*)d_out;

  const size_t ND = (size_t)N_ * D_;
  const size_t ED = (size_t)E_ * D_;

  uint8_t* wp = (uint8_t*)d_ws;
  auto carve = [&](size_t bytes) -> void* {
    void* p = (void*)wp;
    wp += (bytes + 255) & ~(size_t)255;
    return p;
  };

  _Float16* x_h    = (_Float16*)carve(ND * 2);
  float*    xbuf   = (float*)carve(ND * 4);
  float*    qkv    = (float*)carve((size_t)N_ * 3 * D_ * 4);
  _Float16* q_h    = (_Float16*)carve(ND * 2);
  _Float16* k_h    = (_Float16*)carve(ND * 2);
  _Float16* v_h    = (_Float16*)carve(ND * 2);
  float*    obuf   = (float*)carve(ND * 4);
  _Float16* o_h    = (_Float16*)carve(ND * 2);
  float*    y1     = (float*)carve(ND * 4);
  float*    hwv    = (float*)carve((size_t)N_ * 4);
  float*    xl     = (float*)carve(ND * 4);
  float*    e_attr = (float*)carve(ED * 4);
  float*    alpha  = (float*)carve((size_t)NNZ_ * 4);
  float*    exb    = (float*)carve((size_t)NNZ_ * 4);
  float*    deg    = (float*)carve((size_t)E_ * 4);
  float*    Binv   = (float*)carve((size_t)E_ * 4);
  float*    mbuf   = (float*)carve((size_t)N_ * 4);
  float*    ssum   = (float*)carve((size_t)N_ * 4);
  float*    Dw     = (float*)carve((size_t)N_ * 4);
  float*    Dinv   = (float*)carve((size_t)N_ * 4);
  float*    ef     = (float*)carve(ED * 4);
  float*    outb   = (float*)carve(ND * 4);
  float*    h64    = (float*)carve((size_t)N_ * 64 * 4);
  _Float16* h64_h  = (_Float16*)carve((size_t)N_ * 64 * 2);
  float*    mu     = (float*)carve(64 * 4);
  float*    var    = (float*)carve(64 * 4);
  _Float16* inw_h  = (_Float16*)carve((size_t)3 * D_ * D_ * 2);
  _Float16* outw_h = (_Float16*)carve((size_t)D_ * D_ * 2);
  _Float16* convw_h= (_Float16*)carve((size_t)L_ * D_ * D_ * 2);
  _Float16* wgw1_h = (_Float16*)carve((size_t)D_ * D_ * 2);
  _Float16* flw1_h = (_Float16*)carve((size_t)64 * D_ * 2);
  _Float16* flw2_h = (_Float16*)carve((size_t)D_ * 64 * 2);

  // --- convert weights to f16 ---
  cvt_f16<<<nb(3 * D_ * D_, 256), 256, 0, stream>>>(in_w, inw_h, (size_t)3 * D_ * D_);
  cvt_f16<<<nb(D_ * D_, 256), 256, 0, stream>>>(out_w, outw_h, (size_t)D_ * D_);
  cvt_f16<<<nb(L_ * D_ * D_, 256), 256, 0, stream>>>(conv_w, convw_h, (size_t)L_ * D_ * D_);
  cvt_f16<<<nb(D_ * D_, 256), 256, 0, stream>>>(wg_w1, wgw1_h, (size_t)D_ * D_);
  cvt_f16<<<nb(64 * D_, 256), 256, 0, stream>>>(fl_w1, flw1_h, (size_t)64 * D_);
  cvt_f16<<<nb(D_ * 64, 256), 256, 0, stream>>>(fl_w2, flw2_h, (size_t)D_ * 64);

  // --- embedding gather ---
  gather_embed<<<N_, D_, 0, stream>>>(emb, kw, x_h);

  // --- QKV projection + attention ---
  wmma_gemm_bias<<<(N_ / 16) * (3 * D_ / 16), 32, 0, stream>>>(x_h, inw_h, in_b, qkv, N_, 3 * D_, D_, 0);
  split_qkv_h<<<nb(ND, 256), 256, 0, stream>>>(qkv, q_h, k_h, v_h);
  flash_attn_kernel<<<dim3(N_ / 16, H_), 32, 0, stream>>>(q_h, k_h, v_h, obuf);
  cvt_f16<<<nb(ND, 256), 256, 0, stream>>>(obuf, o_h, ND);
  wmma_gemm_bias<<<(N_ / 16) * (D_ / 16), 32, 0, stream>>>(o_h, outw_h, out_b, xbuf, N_, D_, D_, 0);
  cvt_f16<<<nb(ND, 256), 256, 0, stream>>>(xbuf, x_h, ND);

  // --- edge degrees ---
  fill_f32<<<nb(E_, 256), 256, 0, stream>>>(deg, 0.0f, E_);
  compute_deg<<<nb(NNZ_, 256), 256, 0, stream>>>(edge_idx, deg);
  recip_pos<<<nb(E_, 256), 256, 0, stream>>>(deg, Binv, E_);

  // --- hypergraph conv layers ---
  for (int l = 0; l < L_; ++l) {
    wmma_gemm_bias<<<(N_ / 16) * (D_ / 16), 32, 0, stream>>>(x_h, wgw1_h, wg_b1, y1, N_, D_, D_, 1);
    hw_kernel<<<nb((size_t)N_ * 32, 256), 256, 0, stream>>>(y1, wg_w2, wg_b2, hwv);
    wmma_gemm_bias<<<(N_ / 16) * (D_ / 16), 32, 0, stream>>>(x_h, convw_h + (size_t)l * D_ * D_, nullptr, xl, N_, D_, D_, 0);
    fill_f32<<<nb(ED, 256), 256, 0, stream>>>(e_attr, 0.0f, ED);
    scatter_edge_sum<<<NNZ_, D_, 0, stream>>>(xl, node_idx, edge_idx, e_attr);
    scale_rows<<<nb(ED, 256), 256, 0, stream>>>(e_attr, Binv, ED);
    alpha_kernel<<<nb((size_t)NNZ_ * 32, 256), 256, 0, stream>>>(
        xl, e_attr, node_idx, edge_idx, conv_att + (size_t)l * 2 * D_,
        conv_att + (size_t)l * 2 * D_ + D_, alpha);
    fill_f32<<<nb(N_, 256), 256, 0, stream>>>(mbuf, -INFINITY, N_);
    seg_max_kernel<<<nb(NNZ_, 256), 256, 0, stream>>>(alpha, node_idx, mbuf);
    fix_m<<<nb(N_, 256), 256, 0, stream>>>(mbuf, N_);
    fill_f32<<<nb(N_, 256), 256, 0, stream>>>(ssum, 0.0f, N_);
    ex_kernel<<<nb(NNZ_, 256), 256, 0, stream>>>(alpha, mbuf, node_idx, exb, ssum);
    norm_alpha<<<nb(NNZ_, 256), 256, 0, stream>>>(exb, ssum, node_idx, alpha);
    fill_f32<<<nb(N_, 256), 256, 0, stream>>>(Dw, 0.0f, N_);
    dw_kernel<<<nb(NNZ_, 256), 256, 0, stream>>>(hwv, node_idx, edge_idx, Dw);
    recip_pos<<<nb(N_, 256), 256, 0, stream>>>(Dw, Dinv, N_);
    fill_f32<<<nb(ED, 256), 256, 0, stream>>>(ef, 0.0f, ED);
    scatter_msg1<<<NNZ_, D_, 0, stream>>>(xl, alpha, Binv, node_idx, edge_idx, ef);
    fill_f32<<<nb(ND, 256), 256, 0, stream>>>(outb, 0.0f, ND);
    scatter_msg2<<<NNZ_, D_, 0, stream>>>(ef, alpha, Dinv, node_idx, edge_idx, outb);
    finish_layer<<<nb(ND, 256), 256, 0, stream>>>(outb, conv_b + (size_t)l * D_, xbuf, x_h);
  }

  // --- final MLP with batchnorm ---
  wmma_gemm_bias<<<(N_ / 16) * (64 / 16), 32, 0, stream>>>(x_h, flw1_h, fl_b1, h64, N_, 64, D_, 0);
  bn_stats<<<64, 256, 0, stream>>>(h64, mu, var);
  bn_apply<<<nb((size_t)N_ * 64, 256), 256, 0, stream>>>(h64, mu, var, bn_gamma, bn_beta, h64_h);
  wmma_gemm_bias<<<(N_ / 16) * (D_ / 16), 32, 0, stream>>>(h64_h, flw2_h, fl_b2, outp, N_, D_, 64, 0);
}
